// DeformableFusionAcrossFocus_29051158790804
// MI455X (gfx1250) — matrix operand order repaired and checked by reading, not compiled, not run on hardware
//
#include <hip/hip_runtime.h>
#include <hip/hip_bf16.h>

typedef _Float16 v16h __attribute__((ext_vector_type(16)));
typedef _Float16 v8h  __attribute__((ext_vector_type(8)));
typedef float    v8f  __attribute__((ext_vector_type(8)));

// Problem dims (fixed by the reference)
constexpr int Bn = 2, C = 64, N = 16, H = 96, W = 96, Kk = 3;
constexpr int HW = H * W;          // 9216
constexpr int Mtot = Bn * HW;      // 18432 positions
constexpr int R = C * Kk;          // 192 = flattened (c,k) reduction dim
constexpr int TW = 4;              // consecutive w-positions per block (1 per wave)
constexpr int NSTRIP = Mtot / TW;  // 4608
constexpr int THREADS = 128;       // 4 wave32s
constexpr int NQ = R / 32;         // 6 K-chunks of 32 for wmma_f32_16x16x32_f16
constexpr int GRID = 1152;         // 4 strips per block
constexpr int BUFLEN = C * N * TW; // 4096 floats = 16 KB per strip buffer
constexpr int ROWS_PER_THREAD = (C * N) / THREADS;  // 8

// Issue one strip (4 consecutive w positions) as async global->LDS b128 copies.
// Each thread issues exactly ROWS_PER_THREAD (8) async ops (ASYNCcnt += 8).
__device__ __forceinline__ void issue_async_strip(const float* __restrict__ x,
                                                  int s, float* lbuf, int tid)
{
    const int m0 = s * TW;
    const int b  = m0 / HW;
    const int hw = m0 % HW;                        // multiple of 4 -> 16B aligned
    const float* xb = x + (size_t)b * (size_t)(C * N * HW) + hw;
    #pragma unroll
    for (int j = 0; j < ROWS_PER_THREAD; ++j) {
        const int i = tid + THREADS * j;           // (c*16+n) row index
        const unsigned long long ga = (unsigned long long)(uintptr_t)(xb + (size_t)i * HW);
        const unsigned int       la = (unsigned int)(uintptr_t)(lbuf + i * 4); // low32 = LDS offset
        asm volatile("global_load_async_to_lds_b128 %0, %1, off"
                     :: "v"(la), "v"(ga) : "memory");
    }
}

__global__ __launch_bounds__(THREADS)
void deform_fuse_kernel(const float* __restrict__ x,
                        const float* __restrict__ w_off,
                        const float* __restrict__ b_off,
                        const float* __restrict__ w_def,
                        const float* __restrict__ b_def,
                        float* __restrict__ out)
{
    // w_def converted to f16 once per block: sWdef[o*R + r], r = c*3+k
    __shared__ __align__(16) _Float16 sWdef[64 * R];           // 24 KB
    // double-buffered strip; each buffer also doubles as the out-stage for its strip
    __shared__ __align__(16) float    sBuf[2 * BUFLEN];        // 32 KB
    __shared__ _Float16 sWoffH[6 * R];                          // 2.25 KB (f16 offset-conv weights)
    __shared__ float    sOffs[TW][6 * N];                       // per-wave conv offsets
    __shared__ __align__(16) float4   sWtab[TW][Kk * N];        // {c0,c1,w0,w1} per (k,n)
    __shared__ float    sBdef[64];
    __shared__ float    sBoff[8];

    const int tid  = threadIdx.x;
    const int lane = tid & 31;
    const int wv   = tid >> 5;

    // ---- one-time per-block preload (small tensors -> LDS) ----
    for (int i = tid; i < 64 * R; i += THREADS) sWdef[i]  = (_Float16)w_def[i];
    for (int i = tid; i < 6 * R;  i += THREADS) sWoffH[i] = (_Float16)w_off[i];
    if (tid < 64) sBdef[tid] = b_def[tid];
    if (tid < 6)  sBoff[tid] = b_off[tid];

    // ---- prologue: kick off the first strip's async copy into buffer 0 ----
    issue_async_strip(x, blockIdx.x, sBuf, tid);
    int cur = 0;

    __syncthreads();   // preload visibility (async strip is guarded by asynccnt below)

    for (int s = blockIdx.x; s < NSTRIP; s += GRID) {
        // issue next strip into the other buffer (clamped re-issue keeps ASYNCcnt uniform)
        const int sn = (s + GRID < NSTRIP) ? (s + GRID) : s;
        issue_async_strip(x, sn, sBuf + (cur ^ 1) * BUFLEN, tid);
        // prev 8 async ops complete (in-order) once count <= 8
        asm volatile("s_wait_asynccnt 0x8" ::: "memory");
        __syncthreads();

        float* xm = sBuf + cur * BUFLEN;   // this strip: xm[(c*16+n)*4 + p]

        const int m0 = s * TW;
        const int b  = m0 / HW;
        const int hw = m0 % HW;
        float* ob = out + (size_t)b * (size_t)(C * N * HW) + hw;

        const int p = wv;  // this wave's position slot within the strip

        // ---- offsets conv: offs[o][n] = b_off[o] + sum_{c,k} w_off[o,c,k]*xm[c][n-1+k]
        for (int rep = 0; rep < 3; ++rep) {
            const int item = lane + 32 * rep;      // = o*16 + n
            const int o = item >> 4, n = item & 15;
            float acc = sBoff[o];
            for (int c = 0; c < C; ++c) {
                #pragma unroll
                for (int k = 0; k < Kk; ++k) {
                    const int nk = n - 1 + k;
                    if (nk >= 0 && nk < N)
                        acc = fmaf((float)sWoffH[o * R + c * Kk + k],
                                   xm[(c * 16 + nk) * 4 + p], acc);
                }
            }
            sOffs[wv][item] = acc;
        }

        // ---- bilinear gather weights/indices per (k,n): 48 items
        for (int rep = 0; rep < 2; ++rep) {
            const int item = lane + 32 * rep;
            if (item < Kk * N) {
                const int k = item >> 4, n = item & 15;
                const float offy = sOffs[wv][(2 * k) * 16 + n];       // offsets[0::2]
                const float offx = sOffs[wv][(2 * k + 1) * 16 + n];   // offsets[1::2]
                const float px  = (float)(n - 1 + k) + offx;
                const float x0f = floorf(px);
                const float fx  = px - x0f;
                const int   x0i = (int)x0f;
                const int   x1i = x0i + 1;
                const float wy  = fmaxf(0.f, 1.f - fabsf(offy));
                const float w0  = (x0i >= 0 && x0i < N) ? (1.f - fx) * wy : 0.f;
                const float w1  = (x1i >= 0 && x1i < N) ? fx * wy : 0.f;
                const int   c0  = min(max(x0i, 0), N - 1);
                const int   c1  = min(max(x1i, 0), N - 1);
                sWtab[wv][item] = make_float4(__int_as_float(c0), __int_as_float(c1), w0, w1);
            }
        }

        // ---- main GEMM: out(64x16) = Wdef(64x192) x val(192x16), f16 WMMA, f32 acc
        v8f acc[4] = {};                 // 4 o-tiles of 16 rows
        const int n  = lane & 15;        // D/B column
        const int hi = lane >> 4;

        for (int q = 0; q < NQ; ++q) {
            // Build B fragment (32x16, K x N) directly in registers.
            // 16-bit B striping: lanes 0-15 hold K={0..7,16..23}, lanes 16-31 K={8..15,24..31}.
            v16h bfrag;
            #pragma unroll
            for (int sl = 0; sl < 16; ++sl) {
                const int kloc = (sl < 8) ? (hi * 8 + sl) : (16 + hi * 8 + (sl - 8));
                const int r    = q * 32 + kloc;
                const int c    = r / 3;
                const int k    = r - 3 * c;
                const float4 wt = sWtab[wv][k * 16 + n];
                const int ci0 = __float_as_int(wt.x);
                const int ci1 = __float_as_int(wt.y);
                const float g = xm[(c * 16 + ci0) * 4 + p] * wt.z +
                                xm[(c * 16 + ci1) * 4 + p] * wt.w;
                bfrag[sl] = (_Float16)g;
            }
            // A fragments per ISA 16-bit A layout:
            //   lane<16 : K = {0..7, 16..23}; lane>=16 : K = {8..15, 24..31}
            const int arow = lane & 15;
            const int acol = q * 32 + (hi ? 8 : 0);
            #pragma unroll
            for (int t = 0; t < 4; ++t) {
                const _Float16* ap = &sWdef[(t * 16 + arow) * R + acol];
                const v8h alo = *(const v8h*)ap;          // ds_load_b128
                const v8h ahi = *(const v8h*)(ap + 16);   // ds_load_b128
                v16h afrag;
                #pragma unroll
                for (int i = 0; i < 8; ++i) { afrag[i] = alo[i]; afrag[i + 8] = ahi[i]; }
                acc[t] = __builtin_amdgcn_wmma_f32_16x16x32_f16(
                    false, afrag, false, bfrag, (short)0, acc[t], false, false);
            }
        }

        // ---- bias + stage D into this strip's buffer (own p-slots only; disjoint per wave)
        // D layout: VGPR j holds M=j (lanes 0-15) / M=j+8 (lanes 16-31)
        #pragma unroll
        for (int t = 0; t < 4; ++t) {
            #pragma unroll
            for (int j = 0; j < 8; ++j) {
                const int o = t * 16 + j + 8 * hi;
                xm[(o * 16 + n) * 4 + p] = acc[t][j] + sBdef[o];
            }
        }
        __syncthreads();

        // cooperative coalesced strip store (float4 = 4 consecutive w per (o,n))
        for (int i = tid; i < C * N; i += THREADS) {
            const float4 v = *(const float4*)(&xm[i * 4]);
            *(float4*)(ob + (size_t)i * HW) = v;
        }
        cur ^= 1;
        __syncthreads();   // protect this buffer from the next async issue
    }
}

extern "C" void kernel_launch(void* const* d_in, const int* in_sizes, int n_in,
                              void* d_out, int out_size, void* d_ws, size_t ws_size,
                              hipStream_t stream) {
    const float* x     = (const float*)d_in[0];
    const float* w_off = (const float*)d_in[1];
    const float* b_off = (const float*)d_in[2];
    const float* w_def = (const float*)d_in[3];
    const float* b_def = (const float*)d_in[4];
    float* out = (float*)d_out;
    (void)in_sizes; (void)n_in; (void)out_size; (void)d_ws; (void)ws_size;

    hipLaunchKernelGGL(deform_fuse_kernel, dim3(GRID), dim3(THREADS), 0, stream,
                       x, w_off, b_off, w_def, b_def, out);
}
